// RNNSequenceModel_74809740361881
// MI455X (gfx1250) — compile-verified, matrix operand order checked by reading, not run on hardware
//
#include <hip/hip_runtime.h>
#include <hip/hip_bf16.h>
#include <stdint.h>

// ---------------------------------------------------------------------------
// Model dims
// ---------------------------------------------------------------------------
#define Bn   64
#define Sn   256
#define En   1024
#define Hn   1024
#define Gn   4096              // 4*H
#define NCn  4
#define DKn  64
#define BSn  16384             // B*S

typedef __attribute__((ext_vector_type(16))) __bf16 v16bf;
typedef __attribute__((ext_vector_type(8)))  float  v8f;

union Frag { v16bf v; uint4 q[2]; };

__device__ __forceinline__ unsigned short f2bf(float f) {
  unsigned int u = __builtin_bit_cast(unsigned int, f);
  unsigned int r = (u + 0x7FFFu + ((u >> 16) & 1u)) >> 16;   // RNE
  return (unsigned short)r;
}
__device__ __forceinline__ float sigf(float x) { return 1.f / (1.f + __expf(-x)); }
__device__ __forceinline__ float tanhfast(float x) {
  x = fminf(fmaxf(x, -15.f), 15.f);
  float e = __expf(2.f * x);
  return (e - 1.f) / (e + 1.f);
}

// ---------------------------------------------------------------------------
// fp32 -> bf16 convert
// ---------------------------------------------------------------------------
__global__ void cvt_bf16_kernel(const float* __restrict__ src,
                                unsigned short* __restrict__ dst, size_t n) {
  size_t i = (size_t)blockIdx.x * blockDim.x + threadIdx.x;
  size_t st = (size_t)gridDim.x * blockDim.x;
  for (; i < n; i += st) dst[i] = f2bf(src[i]);
}

// ---------------------------------------------------------------------------
// WMMA GEMM:  C[M,N] = A[M,K](bf16) * W[N,K]^T(bf16) + bias1 + bias2
// block 256 thr = 8 waves (2 M-groups x 4 N-groups).
// Wave tile: 32(M) x 16*NSUB(N)  ->  M_BLK=64, N_BLK=64*NSUB.
// Double-buffered LDS A tile: one barrier per K step.
// ---------------------------------------------------------------------------
template <int NSUB>
__global__ __launch_bounds__(256) void gemm_bf16_wmma(
    const unsigned short* __restrict__ A, const unsigned short* __restrict__ Wm,
    const float* __restrict__ bias1, const float* __restrict__ bias2,
    float* __restrict__ C, int M, int N, int K) {
  __shared__ unsigned short As[2][64][32];
  const int tid = threadIdx.x;
  const int lane = tid & 31, wv = tid >> 5;
  const int lh = lane >> 4, l15 = lane & 15;
  const int m0 = blockIdx.y * 64;
  const int n0 = blockIdx.x * (64 * NSUB);
  const int mi = wv & 1, ni = wv >> 1;

  v8f acc[2][NSUB];
  for (int a = 0; a < 2; ++a)
    for (int b = 0; b < NSUB; ++b) acc[a][b] = (v8f)0.f;

  const int arow = tid >> 2;          // 0..63
  const int acol = (tid & 3) * 8;     // 0,8,16,24
  int arg = m0 + arow; if (arg >= M) arg = M - 1;
  const unsigned short* Arow = A + (size_t)arg * K;

  const unsigned short* Wrow[NSUB];
  for (int nt = 0; nt < NSUB; ++nt) {
    int n = n0 + ni * (16 * NSUB) + nt * 16 + l15;
    if (n >= N) n = N - 1;
    Wrow[nt] = Wm + (size_t)n * K;
  }

  const int kiters = K / 32;
  // prologue: stage tile 0
  *(uint4*)&As[0][arow][acol] = *(const uint4*)(Arow + acol);

  for (int ki = 0; ki < kiters; ++ki) {
    const int cur = ki & 1, nxt = cur ^ 1;
    const int k0 = ki * 32;
    // prefetch next A tile into regs (no LDS hazard yet)
    uint4 areg;
    if (ki + 1 < kiters) {
      areg = *(const uint4*)(Arow + k0 + 32 + acol);
      __builtin_prefetch(Wrow[0] + k0 + 64, 0, 3);
      __builtin_prefetch(Wrow[NSUB - 1] + k0 + 64, 0, 3);
    }
    // B fragments straight from global (L2-resident weights)
    Frag bf[NSUB];
    for (int nt = 0; nt < NSUB; ++nt) {
      const unsigned short* p = Wrow[nt] + k0 + lh * 8;
      bf[nt].q[0] = *(const uint4*)(p);
      bf[nt].q[1] = *(const uint4*)(p + 16);
    }
    __syncthreads();                       // As[cur] ready; As[nxt] readers done
    if (ki + 1 < kiters)
      *(uint4*)&As[nxt][arow][acol] = areg;
    Frag af[2];
    for (int mt = 0; mt < 2; ++mt) {
      int r = mi * 32 + mt * 16 + l15;
      af[mt].q[0] = *(const uint4*)&As[cur][r][lh * 8];
      af[mt].q[1] = *(const uint4*)&As[cur][r][16 + lh * 8];
    }
    for (int mt = 0; mt < 2; ++mt)
      for (int nt = 0; nt < NSUB; ++nt)
        acc[mt][nt] = __builtin_amdgcn_wmma_f32_16x16x32_bf16(
            false, af[mt].v, false, bf[nt].v, (short)0, acc[mt][nt], false, false);
  }

  for (int nt = 0; nt < NSUB; ++nt) {
    int n = n0 + ni * (16 * NSUB) + nt * 16 + l15;
    if (n >= N) continue;
    float bsum = (bias1 ? bias1[n] : 0.f) + (bias2 ? bias2[n] : 0.f);
    for (int mt = 0; mt < 2; ++mt) {
      int mbase = m0 + mi * 32 + mt * 16 + (lh ? 8 : 0);
      for (int j = 0; j < 8; ++j) {
        int m = mbase + j;
        if (m < M) C[(size_t)m * N + n] = acc[mt][nt][j] + bsum;
      }
    }
  }
}

// ---------------------------------------------------------------------------
// LSTM recurrence step, both directions of one layer per launch.
// grid = (H/64, B/16, 2dirs); block 256 = 8 waves.
// Each block: gates[16 x 256] = hprev[16x1024] @ whh_slice^T + xp, then cell.
// ---------------------------------------------------------------------------
__global__ __launch_bounds__(256) void lstm_step_kernel(
    const float* __restrict__ xpF, const float* __restrict__ xpR,
    const unsigned short* __restrict__ whhF, const unsigned short* __restrict__ whhR,
    const float* __restrict__ hInF, float* __restrict__ hOutF, float* __restrict__ cF,
    const float* __restrict__ hInR, float* __restrict__ hOutR, float* __restrict__ cR,
    float* hsfF, float* hsfR, int hsf_stride, int hsf_offF, int hsf_offR,
    unsigned short* hsbF, unsigned short* hsbR, int hsb_stride, int hsb_offF, int hsb_offR,
    int t) {
  __shared__ unsigned short Hs[2][16][32];
  __shared__ float Gs[16][256];

  const int dir = blockIdx.z;
  const float* xp          = dir ? xpR  : xpF;
  const unsigned short* whh= dir ? whhR : whhF;
  const float* hIn         = dir ? hInR : hInF;
  float* hOut              = dir ? hOutR : hOutF;
  float* cSt               = dir ? cR   : cF;
  float* hsf               = dir ? hsfR : hsfF;
  unsigned short* hsb      = dir ? hsbR : hsbF;
  const int hsf_off        = dir ? hsf_offR : hsf_offF;
  const int hsb_off        = dir ? hsb_offR : hsb_offF;
  const int te             = dir ? (Sn - 1 - t) : t;

  const int tid = threadIdx.x;
  const int lane = tid & 31, wv = tid >> 5;
  const int lh = lane >> 4, l15 = lane & 15;
  const int mb = blockIdx.y * 16;
  const int n0 = blockIdx.x * 64;
  const int gsec = wv >> 1;                 // gate section i/f/g/o
  const int ncol = n0 + (wv & 1) * 32;      // hidden col base for this wave

  v8f acc[2];
  acc[0] = (v8f)0.f; acc[1] = (v8f)0.f;

  const unsigned short* Wrow[2];
  for (int nt = 0; nt < 2; ++nt) {
    int row = gsec * Hn + ncol + nt * 16 + l15;   // row in (4H x H) whh
    Wrow[nt] = whh + (size_t)row * Hn;
  }

  const int srow = tid >> 4;          // 0..15
  const int scol = (tid & 15) * 2;    // 0..30 step 2
  const float* hRowBase = hIn + (size_t)(mb + srow) * Hn + scol;

  const int kiters = Hn / 32;
  // prologue: stage h tile 0 (convert fp32 -> bf16)
  Hs[0][srow][scol]     = f2bf(hRowBase[0]);
  Hs[0][srow][scol + 1] = f2bf(hRowBase[1]);

  for (int ki = 0; ki < kiters; ++ki) {
    const int cur = ki & 1, nxt = cur ^ 1;
    const int k0 = ki * 32;
    float nf0 = 0.f, nf1 = 0.f;
    if (ki + 1 < kiters) {
      nf0 = hRowBase[k0 + 32];
      nf1 = hRowBase[k0 + 33];
      __builtin_prefetch(Wrow[0] + k0 + 64, 0, 3);
      __builtin_prefetch(Wrow[1] + k0 + 64, 0, 3);
    }
    Frag bf[2];
    for (int nt = 0; nt < 2; ++nt) {
      const unsigned short* p = Wrow[nt] + k0 + lh * 8;
      bf[nt].q[0] = *(const uint4*)(p);
      bf[nt].q[1] = *(const uint4*)(p + 16);
    }
    __syncthreads();
    if (ki + 1 < kiters) {
      Hs[nxt][srow][scol]     = f2bf(nf0);
      Hs[nxt][srow][scol + 1] = f2bf(nf1);
    }
    Frag af;
    af.q[0] = *(const uint4*)&Hs[cur][l15][lh * 8];
    af.q[1] = *(const uint4*)&Hs[cur][l15][16 + lh * 8];
    for (int nt = 0; nt < 2; ++nt)
      acc[nt] = __builtin_amdgcn_wmma_f32_16x16x32_bf16(
          false, af.v, false, bf[nt].v, (short)0, acc[nt], false, false);
  }

  // gates += xp, scatter to LDS
  for (int nt = 0; nt < 2; ++nt) {
    int gcol = gsec * Hn + ncol + nt * 16 + l15;    // 0..4095
    int lcol = wv * 32 + nt * 16 + l15;             // 0..255
    for (int j = 0; j < 8; ++j) {
      int m = j + (lh ? 8 : 0);
      float v = acc[nt][j] +
                xp[((size_t)(mb + m) * Sn + te) * Gn + gcol];
      Gs[m][lcol] = v;
    }
  }
  __syncthreads();

  // cell update: 16 x 64 cells, 256 threads -> 4 each
  for (int cidx = tid; cidx < 16 * 64; cidx += 256) {
    int m = cidx >> 6, hh = cidx & 63;
    float ig = Gs[m][hh];
    float fg = Gs[m][64 + hh];
    float gg = Gs[m][128 + hh];
    float og = Gs[m][192 + hh];
    size_t ci = (size_t)(mb + m) * Hn + n0 + hh;
    float cv = sigf(fg) * cSt[ci] + sigf(ig) * tanhfast(gg);
    cSt[ci] = cv;
    float hv = sigf(og) * tanhfast(cv);
    hOut[ci] = hv;
    size_t so = (size_t)(mb + m) * Sn + te;
    if (hsf) hsf[so * hsf_stride + hsf_off + n0 + hh] = hv;
    if (hsb) hsb[so * hsb_stride + hsb_off + n0 + hh] = f2bf(hv);
  }
}

// ---------------------------------------------------------------------------
// Windowed attention core: 16 heads, dh=4, window=3, sides share code
// side=0: left ctx positions s-3,s-2,s-1 ; side=1: right ctx s+1,s+2,s+3
// OOB rows -> K=bk, V=bv (projection of zero pad)
// ---------------------------------------------------------------------------
__global__ void attn_win_kernel(const float* __restrict__ Qh,
                                const float* __restrict__ Kh,
                                const float* __restrict__ Vh,
                                const float* __restrict__ bk,
                                const float* __restrict__ bv,
                                unsigned short* __restrict__ ctx_bf, int side) {
  int gid = blockIdx.x * blockDim.x + threadIdx.x;
  if (gid >= BSn * 16) return;
  int n = gid >> 4, hd = gid & 15;
  int b = n / Sn, s = n % Sn;
  const float* q = Qh + (size_t)n * DKn + hd * 4;
  float q0 = q[0], q1 = q[1], q2 = q[2], q3 = q[3];
  float lg[3];
  for (int w = 0; w < 3; ++w) {
    int pos = side ? (s + w + 1) : (s + w - 3);
    const float* kp = (pos >= 0 && pos < Sn)
        ? Kh + ((size_t)(b * Sn + pos) * DKn + hd * 4)
        : bk + hd * 4;
    lg[w] = 0.5f * (q0 * kp[0] + q1 * kp[1] + q2 * kp[2] + q3 * kp[3]);
  }
  float mx = fmaxf(lg[0], fmaxf(lg[1], lg[2]));
  float e0 = __expf(lg[0] - mx), e1 = __expf(lg[1] - mx), e2 = __expf(lg[2] - mx);
  float inv = 1.f / (e0 + e1 + e2);
  float aw[3] = {e0 * inv, e1 * inv, e2 * inv};
  float cx0 = 0.f, cx1 = 0.f, cx2 = 0.f, cx3 = 0.f;
  for (int w = 0; w < 3; ++w) {
    int pos = side ? (s + w + 1) : (s + w - 3);
    const float* vp = (pos >= 0 && pos < Sn)
        ? Vh + ((size_t)(b * Sn + pos) * DKn + hd * 4)
        : bv + hd * 4;
    cx0 += aw[w] * vp[0]; cx1 += aw[w] * vp[1];
    cx2 += aw[w] * vp[2]; cx3 += aw[w] * vp[3];
  }
  unsigned short* d = ctx_bf + (size_t)n * DKn + hd * 4;
  d[0] = f2bf(cx0); d[1] = f2bf(cx1); d[2] = f2bf(cx2); d[3] = f2bf(cx3);
}

// ---------------------------------------------------------------------------
// Classifier + log_softmax: one wave32 per position, cat = [out2h|al|ar]
// ---------------------------------------------------------------------------
__global__ __launch_bounds__(256) void classifier_kernel(
    const float* __restrict__ out2h, const float* __restrict__ al,
    const float* __restrict__ ar, const float* __restrict__ wcls,
    const float* __restrict__ bcls, float* __restrict__ dst) {
  int wv = threadIdx.x >> 5, lane = threadIdx.x & 31;
  int n = blockIdx.x * 8 + wv;
  if (n >= BSn) return;
  float a0 = 0.f, a1 = 0.f, a2 = 0.f, a3 = 0.f;
  for (int k = lane; k < 4 * Hn; k += 32) {
    float x;
    if (k < 2048)       x = out2h[(size_t)n * 2048 + k];
    else if (k < 3072)  x = al[(size_t)n * Hn + (k - 2048)];
    else                x = ar[(size_t)n * Hn + (k - 3072)];
    a0 += x * wcls[k];
    a1 += x * wcls[4096 + k];
    a2 += x * wcls[8192 + k];
    a3 += x * wcls[12288 + k];
  }
  for (int off = 16; off; off >>= 1) {
    a0 += __shfl_xor(a0, off);
    a1 += __shfl_xor(a1, off);
    a2 += __shfl_xor(a2, off);
    a3 += __shfl_xor(a3, off);
  }
  if (lane == 0) {
    float l0 = a0 + bcls[0], l1 = a1 + bcls[1], l2 = a2 + bcls[2], l3 = a3 + bcls[3];
    float mx = fmaxf(fmaxf(l0, l1), fmaxf(l2, l3));
    float se = __expf(l0 - mx) + __expf(l1 - mx) + __expf(l2 - mx) + __expf(l3 - mx);
    float lse = mx + __logf(se);
    float* o = dst + (size_t)n * NCn;
    o[0] = l0 - lse; o[1] = l1 - lse; o[2] = l2 - lse; o[3] = l3 - lse;
  }
}

// ---------------------------------------------------------------------------
// Host orchestration
// ---------------------------------------------------------------------------
extern "C" void kernel_launch(void* const* d_in, const int* in_sizes, int n_in,
                              void* d_out, int out_size, void* d_ws, size_t ws_size,
                              hipStream_t stream) {
  (void)in_sizes; (void)n_in; (void)out_size; (void)ws_size;
  const float* inputs   = (const float*)d_in[0];
  const float* w_ih_l0  = (const float*)d_in[2];
  const float* w_hh_l0  = (const float*)d_in[3];
  const float* b_ih_l0  = (const float*)d_in[4];
  const float* b_hh_l0  = (const float*)d_in[5];
  const float* w_ih_l0r = (const float*)d_in[6];
  const float* w_hh_l0r = (const float*)d_in[7];
  const float* b_ih_l0r = (const float*)d_in[8];
  const float* b_hh_l0r = (const float*)d_in[9];
  const float* w_ih_l1  = (const float*)d_in[10];
  const float* w_hh_l1  = (const float*)d_in[11];
  const float* b_ih_l1  = (const float*)d_in[12];
  const float* b_hh_l1  = (const float*)d_in[13];
  const float* w_ih_l1r = (const float*)d_in[14];
  const float* w_hh_l1r = (const float*)d_in[15];
  const float* b_ih_l1r = (const float*)d_in[16];
  const float* b_hh_l1r = (const float*)d_in[17];
  const float* wq       = (const float*)d_in[18];
  const float* bq       = (const float*)d_in[19];
  const float* wk       = (const float*)d_in[20];
  const float* bk       = (const float*)d_in[21];
  const float* wvp      = (const float*)d_in[22];
  const float* bv       = (const float*)d_in[23];
  const float* wo       = (const float*)d_in[24];
  const float* bo       = (const float*)d_in[25];
  const float* w_cls    = (const float*)d_in[26];
  const float* b_cls    = (const float*)d_in[27];
  float* out_logp       = (float*)d_out;

  char* ws = (char*)d_ws;
  size_t off = 0;
  auto alloc = [&](size_t bytes) -> void* {
    void* p = (void*)(ws + off);
    off = (off + bytes + 255) & ~(size_t)255;
    return p;
  };

  unsigned short* in_bf    = (unsigned short*)alloc((size_t)BSn * En * 2);
  unsigned short* wih0_bf  = (unsigned short*)alloc((size_t)Gn * En * 2);
  unsigned short* wih0r_bf = (unsigned short*)alloc((size_t)Gn * En * 2);
  unsigned short* whh0_bf  = (unsigned short*)alloc((size_t)Gn * Hn * 2);
  unsigned short* whh0r_bf = (unsigned short*)alloc((size_t)Gn * Hn * 2);
  unsigned short* whh1_bf  = (unsigned short*)alloc((size_t)Gn * Hn * 2);
  unsigned short* whh1r_bf = (unsigned short*)alloc((size_t)Gn * Hn * 2);
  unsigned short* wih1_bf  = (unsigned short*)alloc((size_t)Gn * 2 * Hn * 2);
  unsigned short* wih1r_bf = (unsigned short*)alloc((size_t)Gn * 2 * Hn * 2);
  unsigned short* wq_bf    = (unsigned short*)alloc((size_t)DKn * Hn * 2);
  unsigned short* wk_bf    = (unsigned short*)alloc((size_t)DKn * Hn * 2);
  unsigned short* wv_bf    = (unsigned short*)alloc((size_t)DKn * Hn * 2);
  unsigned short* wo_bf    = (unsigned short*)alloc((size_t)Hn * DKn * 2);
  float* xpF               = (float*)alloc((size_t)BSn * Gn * 4);
  float* xpR               = (float*)alloc((size_t)BSn * Gn * 4);
  unsigned short* h1_bf    = (unsigned short*)alloc((size_t)BSn * 2 * Hn * 2);
  float* out2h             = (float*)alloc((size_t)BSn * 2 * Hn * 4);
  unsigned short* hl_bf    = (unsigned short*)alloc((size_t)BSn * Hn * 2);
  unsigned short* hr_bf    = (unsigned short*)alloc((size_t)BSn * Hn * 2);
  float* hpFa = (float*)alloc((size_t)Bn * Hn * 4);
  float* hpFb = (float*)alloc((size_t)Bn * Hn * 4);
  float* hpRa = (float*)alloc((size_t)Bn * Hn * 4);
  float* hpRb = (float*)alloc((size_t)Bn * Hn * 4);
  float* cF   = (float*)alloc((size_t)Bn * Hn * 4);
  float* cR   = (float*)alloc((size_t)Bn * Hn * 4);
  float* QhL  = (float*)alloc((size_t)BSn * DKn * 4);
  float* KhL  = (float*)alloc((size_t)BSn * DKn * 4);
  float* VhL  = (float*)alloc((size_t)BSn * DKn * 4);
  float* QhR  = (float*)alloc((size_t)BSn * DKn * 4);
  float* KhR  = (float*)alloc((size_t)BSn * DKn * 4);
  float* VhR  = (float*)alloc((size_t)BSn * DKn * 4);
  unsigned short* ctxL = (unsigned short*)alloc((size_t)BSn * DKn * 2);
  unsigned short* ctxR = (unsigned short*)alloc((size_t)BSn * DKn * 2);
  float* al   = (float*)alloc((size_t)BSn * Hn * 4);
  float* ar   = (float*)alloc((size_t)BSn * Hn * 4);

  auto cvt = [&](const float* src, unsigned short* dst, size_t n) {
    size_t nb = (n + 255) / 256; if (nb > 4096) nb = 4096;
    cvt_bf16_kernel<<<dim3((unsigned)nb), dim3(256), 0, stream>>>(src, dst, n);
  };
  auto gemm = [&](const unsigned short* A, const unsigned short* Wm,
                  const float* b1, const float* b2, float* C, int M, int N, int K) {
    if ((N & 255) == 0) {
      dim3 g(N / 256, (M + 63) / 64);
      gemm_bf16_wmma<4><<<g, dim3(256), 0, stream>>>(A, Wm, b1, b2, C, M, N, K);
    } else {
      dim3 g((N + 127) / 128, (M + 63) / 64);
      gemm_bf16_wmma<2><<<g, dim3(256), 0, stream>>>(A, Wm, b1, b2, C, M, N, K);
    }
  };

  // --- weight / input conversion to bf16 ---
  cvt(inputs,   in_bf,    (size_t)BSn * En);
  cvt(w_ih_l0,  wih0_bf,  (size_t)Gn * En);
  cvt(w_ih_l0r, wih0r_bf, (size_t)Gn * En);
  cvt(w_hh_l0,  whh0_bf,  (size_t)Gn * Hn);
  cvt(w_hh_l0r, whh0r_bf, (size_t)Gn * Hn);
  cvt(w_hh_l1,  whh1_bf,  (size_t)Gn * Hn);
  cvt(w_hh_l1r, whh1r_bf, (size_t)Gn * Hn);
  cvt(w_ih_l1,  wih1_bf,  (size_t)Gn * 2 * Hn);
  cvt(w_ih_l1r, wih1r_bf, (size_t)Gn * 2 * Hn);
  cvt(wq, wq_bf, (size_t)DKn * Hn);
  cvt(wk, wk_bf, (size_t)DKn * Hn);
  cvt(wvp, wv_bf, (size_t)DKn * Hn);
  cvt(wo, wo_bf, (size_t)Hn * DKn);

  // --- layer 0: input projections (both directions) ---
  gemm(in_bf, wih0_bf,  b_ih_l0,  b_hh_l0,  xpF, BSn, Gn, En);
  gemm(in_bf, wih0r_bf, b_ih_l0r, b_hh_l0r, xpR, BSn, Gn, En);

  // --- layer 0 recurrence ---
  (void)hipMemsetAsync(hpFa, 0, (size_t)Bn * Hn * 4, stream);
  (void)hipMemsetAsync(hpRa, 0, (size_t)Bn * Hn * 4, stream);
  (void)hipMemsetAsync(cF,   0, (size_t)Bn * Hn * 4, stream);
  (void)hipMemsetAsync(cR,   0, (size_t)Bn * Hn * 4, stream);
  {
    dim3 g(Hn / 64, Bn / 16, 2);
    float *hiF = hpFa, *hoF = hpFb, *hiR = hpRa, *hoR = hpRb;
    for (int t = 0; t < Sn; ++t) {
      lstm_step_kernel<<<g, dim3(256), 0, stream>>>(
          xpF, xpR, whh0_bf, whh0r_bf,
          hiF, hoF, cF, hiR, hoR, cR,
          nullptr, nullptr, 0, 0, 0,
          h1_bf, h1_bf, 2 * Hn, 0, Hn, t);
      float* tmp = hiF; hiF = hoF; hoF = tmp;
      tmp = hiR; hiR = hoR; hoR = tmp;
    }
  }

  // --- layer 1: input projections (K = 2H) ---
  gemm(h1_bf, wih1_bf,  b_ih_l1,  b_hh_l1,  xpF, BSn, Gn, 2 * Hn);
  gemm(h1_bf, wih1r_bf, b_ih_l1r, b_hh_l1r, xpR, BSn, Gn, 2 * Hn);

  // --- layer 1 recurrence ---
  (void)hipMemsetAsync(hpFa, 0, (size_t)Bn * Hn * 4, stream);
  (void)hipMemsetAsync(hpRa, 0, (size_t)Bn * Hn * 4, stream);
  (void)hipMemsetAsync(cF,   0, (size_t)Bn * Hn * 4, stream);
  (void)hipMemsetAsync(cR,   0, (size_t)Bn * Hn * 4, stream);
  {
    dim3 g(Hn / 64, Bn / 16, 2);
    float *hiF = hpFa, *hoF = hpFb, *hiR = hpRa, *hoR = hpRb;
    for (int t = 0; t < Sn; ++t) {
      lstm_step_kernel<<<g, dim3(256), 0, stream>>>(
          xpF, xpR, whh1_bf, whh1r_bf,
          hiF, hoF, cF, hiR, hoR, cR,
          out2h, out2h, 2 * Hn, 0, Hn,
          hl_bf, hr_bf, Hn, 0, 0, t);
      float* tmp = hiF; hiF = hoF; hoF = tmp;
      tmp = hiR; hiR = hoR; hoR = tmp;
    }
  }

  // --- attention projections (Q/K/V for both sides) ---
  gemm(hl_bf, wq_bf, bq, nullptr, QhL, BSn, DKn, Hn);
  gemm(hl_bf, wk_bf, bk, nullptr, KhL, BSn, DKn, Hn);
  gemm(hl_bf, wv_bf, bv, nullptr, VhL, BSn, DKn, Hn);
  gemm(hr_bf, wq_bf, bq, nullptr, QhR, BSn, DKn, Hn);
  gemm(hr_bf, wk_bf, bk, nullptr, KhR, BSn, DKn, Hn);
  gemm(hr_bf, wv_bf, bv, nullptr, VhR, BSn, DKn, Hn);

  // --- windowed attention cores ---
  {
    int total = BSn * 16;
    dim3 g((total + 255) / 256);
    attn_win_kernel<<<g, dim3(256), 0, stream>>>(QhL, KhL, VhL, bk, bv, ctxL, 0);
    attn_win_kernel<<<g, dim3(256), 0, stream>>>(QhR, KhR, VhR, bk, bv, ctxR, 1);
  }

  // --- attention output projections ---
  gemm(ctxL, wo_bf, bo, nullptr, al, BSn, Hn, DKn);
  gemm(ctxR, wo_bf, bo, nullptr, ar, BSn, Hn, DKn);

  // --- classifier + log_softmax ---
  classifier_kernel<<<dim3(BSn / 8), dim3(256), 0, stream>>>(
      out2h, al, ar, w_cls, b_cls, out_logp);
}
